// Transformer_6064493822348
// MI455X (gfx1250) — matrix operand order repaired and checked by reading, not compiled, not run on hardware
//
#include <hip/hip_runtime.h>
#include <cstdint>
#include <cstddef>

// ---------------------------------------------------------------------------
// Model dimensions (compile-time)
// ---------------------------------------------------------------------------
constexpr int B_SZ    = 2;
constexpr int SEQ     = 2048;
constexpr int DMODEL  = 512;
constexpr int NHEADS  = 8;
constexpr int HDIM    = 64;
constexpr int NLAYERS = 6;
constexpr int FFDIM   = 2048;
constexpr int VOCAB   = 32000;
constexpr int NTOK    = B_SZ * SEQ;           // 4096 rows in the activation GEMMs

typedef __attribute__((ext_vector_type(16))) __bf16 v16bf;
typedef __attribute__((ext_vector_type(8)))  __bf16 v8bf;
typedef __attribute__((ext_vector_type(8)))  float  v8f;
typedef __attribute__((ext_vector_type(4)))  unsigned int v4u;
typedef __attribute__((ext_vector_type(8)))  int v8i;
typedef __attribute__((ext_vector_type(4)))  int v4i;

__device__ __forceinline__ __bf16 f2bf(float f) {
  union { float f; uint32_t u; } a; a.f = f;
  uint32_t r = a.u + 0x7FFFu + ((a.u >> 16) & 1u);   // round-to-nearest-even
  uint16_t h = (uint16_t)(r >> 16);
  return __builtin_bit_cast(__bf16, h);
}

// ---------------------------------------------------------------------------
// CDNA5 async global->LDS copy (ASYNCcnt path, ISA 15.18.3 opcode 98).
// Generic pointers to __shared__ carry the LDS byte offset in the low 32 bits.
// ---------------------------------------------------------------------------
__device__ __forceinline__ void ld_async_b128(void* lds_ptr, const void* gptr) {
  uint32_t l = (uint32_t)(uintptr_t)lds_ptr;
  asm volatile("global_load_async_to_lds_b128 %0, %1, off"
               :: "v"(l), "v"(gptr)
               : "memory");
}
__device__ __forceinline__ void wait_async() {
  asm volatile("s_wait_asynccnt 0x0" ::: "memory");
}

// ---------------------------------------------------------------------------
// f32 -> bf16 conversion (weights, once per launch)
// ---------------------------------------------------------------------------
__global__ void cvt_kernel(const float* __restrict__ in, __bf16* __restrict__ out, long n) {
  long i = (long)blockIdx.x * 256 + threadIdx.x;
  if (i < n) out[i] = f2bf(in[i]);
}

// ---------------------------------------------------------------------------
// Embedding gather: x[tok, d] = emb[tokens[tok], d]   (f32 residual stream)
// ---------------------------------------------------------------------------
__global__ void embed_kernel(const int* __restrict__ tok, const float* __restrict__ emb,
                             float* __restrict__ x) {
  int i = blockIdx.x * 256 + threadIdx.x;
  if (i >= NTOK * DMODEL) return;
  int t = i >> 9;                 // DMODEL = 512
  int d = i & (DMODEL - 1);
  x[i] = emb[(size_t)tok[t] * DMODEL + d];
}

// ---------------------------------------------------------------------------
// LayerNorm (one block per row of 512) -> bf16 output
// ---------------------------------------------------------------------------
__global__ __launch_bounds__(256) void ln_kernel(const float* __restrict__ x,
                                                 const float* __restrict__ g,
                                                 const float* __restrict__ b,
                                                 __bf16* __restrict__ out) {
  __shared__ float red[256];
  const int row = blockIdx.x, tid = threadIdx.x;
  const float* xr = x + (size_t)row * DMODEL;
  float s = 0.0f;
  for (int i = tid; i < DMODEL; i += 256) s += xr[i];
  red[tid] = s; __syncthreads();
  for (int o = 128; o > 0; o >>= 1) { if (tid < o) red[tid] += red[tid + o]; __syncthreads(); }
  const float mu = red[0] * (1.0f / DMODEL);
  __syncthreads();
  float v = 0.0f;
  for (int i = tid; i < DMODEL; i += 256) { float d = xr[i] - mu; v += d * d; }
  red[tid] = v; __syncthreads();
  for (int o = 128; o > 0; o >>= 1) { if (tid < o) red[tid] += red[tid + o]; __syncthreads(); }
  const float rstd = rsqrtf(red[0] * (1.0f / DMODEL) + 1e-5f);
  __bf16* orow = out + (size_t)row * DMODEL;
  for (int i = tid; i < DMODEL; i += 256)
    orow[i] = f2bf((xr[i] - mu) * rstd * g[i] + b[i]);
}

// ---------------------------------------------------------------------------
// Final RMSNorm (eps = float32 machine eps per reference) -> bf16
// ---------------------------------------------------------------------------
__global__ __launch_bounds__(256) void rmsnorm_kernel(const float* __restrict__ x,
                                                      const float* __restrict__ g,
                                                      __bf16* __restrict__ out) {
  __shared__ float red[256];
  const int row = blockIdx.x, tid = threadIdx.x;
  const float* xr = x + (size_t)row * DMODEL;
  float s = 0.0f;
  for (int i = tid; i < DMODEL; i += 256) s += xr[i] * xr[i];
  red[tid] = s; __syncthreads();
  for (int o = 128; o > 0; o >>= 1) { if (tid < o) red[tid] += red[tid + o]; __syncthreads(); }
  const float rms = rsqrtf(red[0] * (1.0f / DMODEL) + 1.1920929e-7f);
  __bf16* orow = out + (size_t)row * DMODEL;
  for (int i = tid; i < DMODEL; i += 256) orow[i] = f2bf(xr[i] * rms * g[i]);
}

// ---------------------------------------------------------------------------
// RoPE + split qkv (f32) -> per-head bf16 Q/K/V, layout [(b*H+h)*SEQ + n, 64]
// ---------------------------------------------------------------------------
__global__ void rope_split_kernel(const float* __restrict__ qkv,
                                  __bf16* __restrict__ qb, __bf16* __restrict__ kb,
                                  __bf16* __restrict__ vb) {
  int idx = blockIdx.x * 256 + threadIdx.x;
  if (idx >= NTOK * NHEADS * (HDIM / 2)) return;
  int j   = idx & 31;                 // pair index 0..31
  int t   = idx >> 5;                 // tok*NHEADS + h
  int h   = t & (NHEADS - 1);
  int tok = t >> 3;
  int n   = tok & (SEQ - 1);
  int b   = tok >> 11;                // SEQ = 2048
  const float* base = qkv + (size_t)tok * (3 * NHEADS * HDIM);
  float freq = __expf(-((float)(2 * j) * (1.0f / HDIM)) * 9.2103403719761836f);
  float ang  = (float)n * freq;
  float sn, cs; __sincosf(ang, &sn, &cs);
  int off = h * HDIM + 2 * j;
  float q0 = base[off],                  q1 = base[off + 1];
  float k0 = base[NHEADS*HDIM + off],    k1 = base[NHEADS*HDIM + off + 1];
  float v0 = base[2*NHEADS*HDIM + off],  v1 = base[2*NHEADS*HDIM + off + 1];
  size_t orow = ((size_t)(b * NHEADS + h) * SEQ + n) * HDIM + 2 * j;
  qb[orow]     = f2bf(q0 * cs - q1 * sn);
  qb[orow + 1] = f2bf(q1 * cs + q0 * sn);
  kb[orow]     = f2bf(k0 * cs - k1 * sn);
  kb[orow + 1] = f2bf(k1 * cs + k0 * sn);
  vb[orow]     = f2bf(v0);
  vb[orow + 1] = f2bf(v1);
}

// ---------------------------------------------------------------------------
// bf16 WMMA GEMM: C[M,N] = epilogue(A[M,K] @ B[K,N])
//   128x128 block tile, BK=32, 8 waves (4 M x 2 N), wave tile 32x64:
//   2 A-fragments x 4 B-fragments -> 8 WMMAs per K-step with fragment reuse.
//   Double-buffered LDS (48KB).
//   A tile: Tensor Data Mover (TENSOR_LOAD_TO_LDS) issued by wave 0 — a D#
//     describing a 128x32-elem 2D tile of the [M,K] bf16 tensor with 8 DWORDs
//     of LDS padding per 64B row (matches the stride-48 LDS layout).
//   B tile: staged transposed ([n][k]) through VGPRs so each B-fragment is a
//     pair of 16B ds_load_b128 per lane.
// Compile-time epilogue: +bias, exact GELU, +residual, f32/bf16 stores.
// ---------------------------------------------------------------------------
template <bool HAS_BIAS, bool HAS_RESID, bool DO_GELU, bool OUTF, bool OUTB>
__global__ __launch_bounds__(256) void gemm_bf16_kernel(
    const __bf16* __restrict__ A, const __bf16* __restrict__ Bw,
    const float* __restrict__ bias, const float* resid,
    float* outF, __bf16* outB, int M, int N, int K)
{
  __shared__ __bf16 Asm[2][128 * 48];   // [m][k], stride 48 (96B, b128-aligned)
  __shared__ __bf16 Bsm[2][128 * 48];   // [n][k], stride 48

  const int tid  = threadIdx.x;
  const int lane = tid & 31, w = tid >> 5;
  const int wm   = w & 3,    wn = w >> 2;
  const int half = lane >> 4, l15 = lane & 15;
  const int m0 = blockIdx.y * 128, n0 = blockIdx.x * 128;

  const int bfk = tid >> 3;             // B fill: 8 threads/k-row, 16 n each
  const int bfn = (tid & 7) * 16;

  v8f acc[2][4] = {};

  // TDM descriptor-driven fill of one A tile (wave 0 only; EXEC is ignored by
  // TDM, issue is per-wave). D# fields per ISA 8.3-8.5.
  auto fillA = [&](int buf, int k0) {
    if (w == 0) {
      const __bf16* g = A + (size_t)m0 * K + k0;        // tile start
      uint64_t ga = (uint64_t)(uintptr_t)g;
      uint32_t la = (uint32_t)(uintptr_t)(void*)&Asm[buf][0];
      v4u g0;
      g0[0] = 1u;                                        // count=1, load, user D#
      g0[1] = la;                                        // lds_addr (bytes)
      g0[2] = (uint32_t)ga;                              // global_addr[31:0]
      g0[3] = (uint32_t)(ga >> 32) | (2u << 30);         // global_addr[56:32] | type=2
      v8i g1;
      g1[0] = (int)((1u << 16)      // data_size = 1 (2 bytes)
                  | (1u << 20)      // pad_enable
                  | (3u << 22)      // pad_interval: pad after 16 DWORDs (64B row)
                  | (7u << 25));    // pad_amount: 8 DWORDs (16 elems) -> stride 48
      g1[1] = (int)(((uint32_t)K & 0xffffu) << 16);                         // tensor_dim0[15:0]
      g1[2] = (int)((((uint32_t)K >> 16) & 0xffffu)
                  | (((uint32_t)M & 0xffffu) << 16));                        // dim0 hi | dim1 lo
      g1[3] = (int)((((uint32_t)M >> 16) & 0xffffu) | (32u << 16));          // dim1 hi | tile_dim0=32
      g1[4] = 128;                                                           // tile_dim1=128
      g1[5] = K;                                                             // tensor_dim0_stride lo
      g1[6] = 0;
      g1[7] = 0;
      v4i gz4 = {0, 0, 0, 0};
      v8i gz8 = {0, 0, 0, 0, 0, 0, 0, 0};
      __builtin_amdgcn_tensor_load_to_lds(g0, g1, gz4, gz4, gz8, 0);
    }
  };
  auto waitA = [&]() {
    if (w == 0) __builtin_amdgcn_s_wait_tensorcnt((short)0);
  };
  auto loadB = [&](int k0, v8bf& r0, v8bf& r1) {
    const __bf16* g = Bw + (size_t)(k0 + bfk) * N + n0 + bfn;
    r0 = *(const v8bf*)g;
    r1 = *(const v8bf*)(g + 8);
  };
  auto storeB = [&](int buf, v8bf r0, v8bf r1) {
    __bf16* base = &Bsm[buf][bfk];
#pragma unroll
    for (int j = 0; j < 8; ++j) base[(bfn + j) * 48] = r0[j];
#pragma unroll
    for (int j = 0; j < 8; ++j) base[(bfn + 8 + j) * 48] = r1[j];
  };

  { // prologue: stage tile 0
    v8bf r0, r1;
    fillA(0, 0);
    loadB(0, r0, r1);
    storeB(0, r0, r1);
  }
  waitA();
  __syncthreads();

  int buf = 0;
  for (int k0 = 0; k0 < K; k0 += 32) {
    const int nxt = buf ^ 1;
    const bool has_next = (k0 + 32) < K;
    v8bf r0, r1;
    if (has_next) { fillA(nxt, k0 + 32); loadB(k0 + 32, r0, r1); }

    union { v16bf v; v8bf h[2]; } a[2], bfr[4];
    const __bf16* Ab = Asm[buf];
    const __bf16* Bb = Bsm[buf];
    const int ar0 = wm * 32 + l15;
    a[0].h[0] = *(const v8bf*)&Ab[ar0 * 48 + half * 8];
    a[0].h[1] = *(const v8bf*)&Ab[ar0 * 48 + 16 + half * 8];
    a[1].h[0] = *(const v8bf*)&Ab[(ar0 + 16) * 48 + half * 8];
    a[1].h[1] = *(const v8bf*)&Ab[(ar0 + 16) * 48 + 16 + half * 8];
#pragma unroll
    for (int j = 0; j < 4; ++j) {
      const int br = wn * 64 + j * 16 + l15;
      bfr[j].h[0] = *(const v8bf*)&Bb[br * 48 + half * 8];
      bfr[j].h[1] = *(const v8bf*)&Bb[br * 48 + 16 + half * 8];
    }
#pragma unroll
    for (int i = 0; i < 2; ++i)
#pragma unroll
      for (int j = 0; j < 4; ++j)
        acc[i][j] = __builtin_amdgcn_wmma_f32_16x16x32_bf16(
            false, a[i].v, false, bfr[j].v, (short)0, acc[i][j], false, false);

    if (has_next) storeB(nxt, r0, r1);
    waitA();
    __syncthreads();
    buf = nxt;
  }

  // straight-line epilogue (all flags compile-time)
#pragma unroll
  for (int j = 0; j < 4; ++j) {
    const int col = n0 + wn * 64 + j * 16 + l15;
    float bval = 0.0f;
    if (HAS_BIAS) bval = bias[col];
#pragma unroll
    for (int i = 0; i < 2; ++i) {
#pragma unroll
      for (int r = 0; r < 8; ++r) {
        int row = m0 + wm * 32 + i * 16 + half * 8 + r;
        float v = acc[i][j][r] + bval;
        if (DO_GELU) v = 0.5f * v * (1.0f + erff(v * 0.70710678118f));
        size_t idx = (size_t)row * N + col;
        if (HAS_RESID) v += resid[idx];
        if (OUTF) outF[idx] = v;
        if (OUTB) outB[idx] = f2bf(v);
      }
    }
  }
}

// ---------------------------------------------------------------------------
// Flash attention, causal, one (b,h) per blockIdx.y, 128 query rows per block.
// 8 waves; each wave owns 16 query rows and keeps m/l stats + 16x64 O in regs.
// K tile filled via async-to-LDS; V tile transposed through VGPRs.
// ---------------------------------------------------------------------------
__global__ __launch_bounds__(256) void attn_kernel(
    const __bf16* __restrict__ Q, const __bf16* __restrict__ Kmat,
    const __bf16* __restrict__ V, __bf16* __restrict__ O)
{
  __shared__ __bf16 Ksm[32 * 64];      // [key][dh]
  __shared__ __bf16 Vts[64 * 32];      // [dh][key] (transposed)
  __shared__ __bf16 Psm[8 * 16 * 32];  // per-wave P re-layout buffer

  const int tid  = threadIdx.x, lane = tid & 31, w = tid >> 5;
  const int half = lane >> 4,   l15  = lane & 15;
  const int bh   = blockIdx.y;
  const int b    = bh >> 3, h = bh & 7;
  const __bf16* Qb = Q    + (size_t)bh * SEQ * HDIM;
  const __bf16* Kb = Kmat + (size_t)bh * SEQ * HDIM;
  const __bf16* Vb = V    + (size_t)bh * SEQ * HDIM;

  const int q0 = blockIdx.x * 128;
  const int qw = q0 + w * 16;          // this wave's query-row base

  // Q fragments (dh 0..31 and 32..63), loaded once
  union { v16bf v; v8bf hh[2]; } qa0, qa1;
  {
    const __bf16* qr = Qb + (size_t)(qw + l15) * HDIM;
    qa0.hh[0] = *(const v8bf*)(qr + half * 8);
    qa0.hh[1] = *(const v8bf*)(qr + 16 + half * 8);
    qa1.hh[0] = *(const v8bf*)(qr + 32 + half * 8);
    qa1.hh[1] = *(const v8bf*)(qr + 48 + half * 8);
  }

  v8f o0 = {}, o1 = {}, o2 = {}, o3 = {};
  float mst[8], lst[8];
#pragma unroll
  for (int r = 0; r < 8; ++r) { mst[r] = -1e30f; lst[r] = 0.0f; }

  const int kmax = q0 + 127;
  __bf16* pw = Psm + w * (16 * 32);

  for (int kb = 0; kb <= kmax; kb += 32) {
    __syncthreads();
    {   // cooperative K/V tile fill (32 keys x 64 dh): K async, V transposed
      int kr = tid >> 3, kc = (tid & 7) * 8;
      ld_async_b128(&Ksm[kr * 64 + kc], &Kb[(size_t)(kb + kr) * HDIM + kc]);
      v8bf vv = *(const v8bf*)&Vb[(size_t)(kb + kr) * HDIM + kc];
#pragma unroll
      for (int j = 0; j < 8; ++j) Vts[(kc + j) * 32 + kr] = vv[j];
    }
    wait_async();
    __syncthreads();
    if (kb > qw + 15) continue;        // tile fully in the causal-masked future

    // ---- S = Q K^T for 32 keys (two 16-key column groups) ----
    v8f sa = {}, sb = {};
    union { v16bf v; v8bf hh[2]; } bb;
    bb.hh[0] = *(const v8bf*)&Ksm[l15 * 64 + half * 8];
    bb.hh[1] = *(const v8bf*)&Ksm[l15 * 64 + 16 + half * 8];
    sa = __builtin_amdgcn_wmma_f32_16x16x32_bf16(false, qa0.v, false, bb.v, (short)0, sa, false, false);
    bb.hh[0] = *(const v8bf*)&Ksm[l15 * 64 + 32 + half * 8];
    bb.hh[1] = *(const v8bf*)&Ksm[l15 * 64 + 48 + half * 8];
    sa = __builtin_amdgcn_wmma_f32_16x16x32_bf16(false, qa1.v, false, bb.v, (short)0, sa, false, false);
    bb.hh[0] = *(const v8bf*)&Ksm[(16 + l15) * 64 + half * 8];
    bb.hh[1] = *(const v8bf*)&Ksm[(16 + l15) * 64 + 16 + half * 8];
    sb = __builtin_amdgcn_wmma_f32_16x16x32_bf16(false, qa0.v, false, bb.v, (short)0, sb, false, false);
    bb.hh[0] = *(const v8bf*)&Ksm[(16 + l15) * 64 + 32 + half * 8];
    bb.hh[1] = *(const v8bf*)&Ksm[(16 + l15) * 64 + 48 + half * 8];
    sb = __builtin_amdgcn_wmma_f32_16x16x32_bf16(false, qa1.v, false, bb.v, (short)0, sb, false, false);

    // ---- online softmax (rows r + 8*half, column = l15 / 16+l15) ----
#pragma unroll
    for (int r = 0; r < 8; ++r) {
      const int qg = qw + half * 8 + r;
      float va = sa[r] * 0.125f;                    // SCALE = DH^-0.5
      float vb2 = sb[r] * 0.125f;
      if (kb + l15 > qg)      va  = -1e30f;
      if (kb + 16 + l15 > qg) vb2 = -1e30f;
      float mx = fmaxf(va, vb2);
      mx = fmaxf(mx, __shfl_xor(mx, 1, 16));
      mx = fmaxf(mx, __shfl_xor(mx, 2, 16));
      mx = fmaxf(mx, __shfl_xor(mx, 4, 16));
      mx = fmaxf(mx, __shfl_xor(mx, 8, 16));
      float mnew = fmaxf(mst[r], mx);
      float fac  = __expf(mst[r] - mnew);
      float pa   = __expf(va - mnew);
      float pb   = __expf(vb2 - mnew);
      float rs = pa + pb;
      rs += __shfl_xor(rs, 1, 16);
      rs += __shfl_xor(rs, 2, 16);
      rs += __shfl_xor(rs, 4, 16);
      rs += __shfl_xor(rs, 8, 16);
      lst[r] = lst[r] * fac + rs;
      mst[r] = mnew;
      o0[r] = o0[r] * fac; o1[r] = o1[r] * fac;
      o2[r] = o2[r] * fac; o3[r] = o3[r] * fac;
      int prow = half * 8 + r;
      pw[prow * 32 + l15]      = f2bf(pa);          // C-layout -> LDS
      pw[prow * 32 + 16 + l15] = f2bf(pb);
    }

    // ---- re-load P as an A-fragment (wave-private region, DScnt-ordered) ----
    union { v16bf v; v8bf hh[2]; } pa_;
    pa_.hh[0] = *(const v8bf*)&pw[l15 * 32 + half * 8];
    pa_.hh[1] = *(const v8bf*)&pw[l15 * 32 + 16 + half * 8];

    // ---- O += P @ V (four 16-col dh groups) ----
    union { v16bf v; v8bf hh[2]; } vb_;
    vb_.hh[0] = *(const v8bf*)&Vts[(0 * 16 + l15) * 32 + half * 8];
    vb_.hh[1] = *(const v8bf*)&Vts[(0 * 16 + l15) * 32 + 16 + half * 8];
    o0 = __builtin_amdgcn_wmma_f32_16x16x32_bf16(false, pa_.v, false, vb_.v, (short)0, o0, false, false);
    vb_.hh[0] = *(const v8bf*)&Vts[(1 * 16 + l15) * 32 + half * 8];
    vb_.hh[1] = *(const v8bf*)&Vts[(1 * 16 + l15) * 32 + 16 + half * 8];
    o1 = __builtin_amdgcn_wmma_f32_16x16x32_bf16(false, pa_.v, false, vb_.v, (short)0, o1, false, false);
    vb_.hh[0] = *(const v8bf*)&Vts[(2 * 16 + l15) * 32 + half * 8];
    vb_.hh[1] = *(const v8bf*)&Vts[(2 * 16 + l15) * 32 + 16 + half * 8];
    o2 = __builtin_amdgcn_wmma_f32_16x16x32_bf16(false, pa_.v, false, vb_.v, (short)0, o2, false, false);
    vb_.hh[0] = *(const v8bf*)&Vts[(3 * 16 + l15) * 32 + half * 8];
    vb_.hh[1] = *(const v8bf*)&Vts[(3 * 16 + l15) * 32 + 16 + half * 8];
    o3 = __builtin_amdgcn_wmma_f32_16x16x32_bf16(false, pa_.v, false, vb_.v, (short)0, o3, false, false);
  }

  // ---- normalize and write O as bf16, layout [b*SEQ + q][h*64 + dh] ----
#pragma unroll
  for (int r = 0; r < 8; ++r) {
    float inv = 1.0f / lst[r];
    int q = qw + half * 8 + r;
    size_t rowoff = ((size_t)(b * SEQ + q)) * (NHEADS * HDIM) + (size_t)h * HDIM;
    O[rowoff + 0  + l15] = f2bf(o0[r] * inv);
    O[rowoff + 16 + l15] = f2bf(o1[r] * inv);
    O[rowoff + 32 + l15] = f2bf(o2[r] * inv);
    O[rowoff + 48 + l15] = f2bf(o3[r] * inv);
  }
}

// ---------------------------------------------------------------------------
// Host-side orchestration
// ---------------------------------------------------------------------------
extern "C" void kernel_launch(void* const* d_in, const int* in_sizes, int n_in,
                              void* d_out, int out_size, void* d_ws, size_t ws_size,
                              hipStream_t stream)
{
  (void)in_sizes; (void)n_in; (void)out_size; (void)ws_size;

  const int*   tokens = (const int*)  d_in[0];
  const float* emb    = (const float*)d_in[1];
  const float* ln1_g  = (const float*)d_in[2];
  const float* ln1_b  = (const float*)d_in[3];
  const float* wqkv   = (const float*)d_in[4];
  const float* wout   = (const float*)d_in[5];
  const float* ln2_g  = (const float*)d_in[6];
  const float* ln2_b  = (const float*)d_in[7];
  const float* w1     = (const float*)d_in[8];
  const float* b1     = (const float*)d_in[9];
  const float* w2     = (const float*)d_in[10];
  const float* b2     = (const float*)d_in[11];
  const float* fing   = (const float*)d_in[12];
  const float* wlog   = (const float*)d_in[13];

  char* ws = (char*)d_ws;
  size_t off = 0;
  auto take = [&](size_t nbytes) -> void* {
    void* p = ws + off;
    off = (off + nbytes + 255) & ~(size_t)255;
    return p;
  };

  const size_t n_wqkv = (size_t)NLAYERS * DMODEL * 3 * NHEADS * HDIM;
  const size_t n_wout = (size_t)NLAYERS * NHEADS * HDIM * DMODEL;
  const size_t n_w1   = (size_t)NLAYERS * DMODEL * FFDIM;
  const size_t n_w2   = (size_t)NLAYERS * FFDIM * DMODEL;
  const size_t n_wlog = (size_t)DMODEL * VOCAB;

  __bf16* wqkv_bf = (__bf16*)take(n_wqkv * 2);
  __bf16* wout_bf = (__bf16*)take(n_wout * 2);
  __bf16* w1_bf   = (__bf16*)take(n_w1 * 2);
  __bf16* w2_bf   = (__bf16*)take(n_w2 * 2);
  __bf16* wlog_bf = (__bf16*)take(n_wlog * 2);
  float*  xbuf    = (float*) take((size_t)NTOK * DMODEL * 4);
  __bf16* hbf     = (__bf16*)take((size_t)NTOK * DMODEL * 2);
  float*  qkvbuf  = (float*) take((size_t)NTOK * 3 * NHEADS * HDIM * 4);
  __bf16* qb      = (__bf16*)take((size_t)B_SZ * NHEADS * SEQ * HDIM * 2);
  __bf16* kb      = (__bf16*)take((size_t)B_SZ * NHEADS * SEQ * HDIM * 2);
  __bf16* vb      = (__bf16*)take((size_t)B_SZ * NHEADS * SEQ * HDIM * 2);
  __bf16* obf     = (__bf16*)take((size_t)NTOK * DMODEL * 2);
  __bf16* g1bf    = (__bf16*)take((size_t)NTOK * FFDIM * 2);

  auto cvt = [&](const float* s, __bf16* d, size_t n) {
    cvt_kernel<<<dim3((unsigned)((n + 255) / 256)), dim3(256), 0, stream>>>(s, d, (long)n);
  };
  cvt(wqkv, wqkv_bf, n_wqkv);
  cvt(wout, wout_bf, n_wout);
  cvt(w1,   w1_bf,   n_w1);
  cvt(w2,   w2_bf,   n_w2);
  cvt(wlog, wlog_bf, n_wlog);

  embed_kernel<<<dim3((NTOK * DMODEL + 255) / 256), dim3(256), 0, stream>>>(tokens, emb, xbuf);

  for (int l = 0; l < NLAYERS; ++l) {
    ln_kernel<<<dim3(NTOK), dim3(256), 0, stream>>>(xbuf, ln1_g + l * DMODEL,
                                                    ln1_b + l * DMODEL, hbf);
    // qkv = h @ Wqkv                       (plain f32 out)
    gemm_bf16_kernel<false, false, false, true, false>
        <<<dim3((3 * NHEADS * HDIM) / 128, NTOK / 128), dim3(256), 0, stream>>>(
            hbf, wqkv_bf + (size_t)l * DMODEL * 3 * NHEADS * HDIM,
            nullptr, nullptr, qkvbuf, nullptr, NTOK, 3 * NHEADS * HDIM, DMODEL);
    rope_split_kernel<<<dim3((NTOK * NHEADS * (HDIM / 2) + 255) / 256), dim3(256), 0, stream>>>(
        qkvbuf, qb, kb, vb);
    attn_kernel<<<dim3(SEQ / 128, B_SZ * NHEADS), dim3(256), 0, stream>>>(qb, kb, vb, obf);
    // x = o @ Wout + x                     (residual, f32 out)
    gemm_bf16_kernel<false, true, false, true, false>
        <<<dim3(DMODEL / 128, NTOK / 128), dim3(256), 0, stream>>>(
            obf, wout_bf + (size_t)l * NHEADS * HDIM * DMODEL,
            nullptr, xbuf, xbuf, nullptr, NTOK, DMODEL, NHEADS * HDIM);
    ln_kernel<<<dim3(NTOK), dim3(256), 0, stream>>>(xbuf, ln2_g + l * DMODEL,
                                                    ln2_b + l * DMODEL, hbf);
    // g1 = gelu(h @ W1 + b1)               (bias+gelu, bf16 out)
    gemm_bf16_kernel<true, false, true, false, true>
        <<<dim3(FFDIM / 128, NTOK / 128), dim3(256), 0, stream>>>(
            hbf, w1_bf + (size_t)l * DMODEL * FFDIM,
            b1 + (size_t)l * FFDIM, nullptr, nullptr, g1bf, NTOK, FFDIM, DMODEL);
    // x = g1 @ W2 + b2 + x                 (bias+residual, f32 out)
    gemm_bf16_kernel<true, true, false, true, false>
        <<<dim3(DMODEL / 128, NTOK / 128), dim3(256), 0, stream>>>(
            g1bf, w2_bf + (size_t)l * FFDIM * DMODEL,
            b2 + (size_t)l * DMODEL, xbuf, xbuf, nullptr, NTOK, DMODEL, FFDIM);
  }

  rmsnorm_kernel<<<dim3(NTOK), dim3(256), 0, stream>>>(xbuf, fing, hbf);
  // logits = xn @ Wlogits                  (plain f32 out, straight to d_out)
  gemm_bf16_kernel<false, false, false, true, false>
      <<<dim3(VOCAB / 128, NTOK / 128), dim3(256), 0, stream>>>(
          hbf, wlog_bf, nullptr, nullptr, (float*)d_out, nullptr, NTOK, VOCAB, DMODEL);
}
